// MFE_22179211117151
// MI455X (gfx1250) — matrix operand order repaired and checked by reading, not compiled, not run on hardware
//
#include <hip/hip_runtime.h>
#include <hip/hip_bf16.h>
#include <math.h>

// ---------------------------------------------------------------------------
// Factorized space-time video transformer forward, CDNA5 (gfx1250, wave32).
// All large GEMMs (~97% of the 250 GFLOP) run through v_wmma_f32_16x16x32_f16
// with f32 accumulation. Weights pre-transposed to K-contiguous f16 so both
// WMMA fragments are contiguous b128 loads in the same k order (k = 16*half + s) for A and B.
// Per-wave register tile is 32x64 (8 WMMAs per 12 b128 loads per K-step) to
// keep the matrix pipe fed from L2-resident weights. Attention (8x8 and
// 196x196, <3% of FLOPs) runs in LDS with scalar VALU.
// ---------------------------------------------------------------------------

#define B_     2
#define F_     8
#define C_     3
#define H_     224
#define W_     224
#define P_     16
#define HP_    14
#define WP_    14
#define NTOK   196                 // HP_*WP_
#define DIM_   768
#define HEADS_ 12
#define DHEAD_ 64
#define INNER_ 768                 // HEADS_*DHEAD_
#define PD_    768                 // C_*P_*P_
#define DEPTH_ 4
#define TOK_   (B_ * F_ * NTOK)    // 3136
#define SCALE_ 0.125f              // DIM_HEAD^-0.5
#define PI_    3.14159265358979323846f
#define NEGMAX (-3.402823466e+38f)

typedef __attribute__((ext_vector_type(16))) _Float16 v16h;
typedef __attribute__((ext_vector_type(8)))  _Float16 v8h;
typedef __attribute__((ext_vector_type(8)))  float    v8f;

// ---------------------------------------------------------------------------
// WMMA GEMM:  C[M,N] = A[M,K] (f16, row-major) x Bt[N,K] (f16, K-contiguous)
// mode 0: C = A*B (+bias)              -> f32 out
// mode 1: C16 = gelu_exact(A*B + bias) -> f16 out (feeds next GEMM)
// mode 2: C += A*B + bias              -> residual accumulate, f32
// Block = 128 threads = 4 waves in a 2(M) x 2(N) grid; each wave computes a
// 32x64 register tile: 2 A fragments x 4 B fragments -> 8 WMMAs per K-step.
// Requires M%64==0, N%128==0, K%32==0 (true for every GEMM in this network).
// ---------------------------------------------------------------------------
__global__ __launch_bounds__(128) void gemm_wmma_kernel(
    const _Float16* __restrict__ A, const _Float16* __restrict__ Bt,
    const float* __restrict__ bias, float* __restrict__ C,
    _Float16* __restrict__ C16, int M, int N, int K, int mode)
{
    const int wave  = threadIdx.x >> 5;   // 0..3
    const int wm    = wave & 1;           // wave row in block (M)
    const int wn    = wave >> 1;          // wave col in block (N)
    const int lane  = threadIdx.x & 31;
    const int lhalf = lane >> 4;          // 0: lanes 0-15, 1: lanes 16-31
    const int lmod  = lane & 15;
    const int m0 = blockIdx.x * 64  + wm * 32;
    const int n0 = blockIdx.y * 128 + wn * 64;

    // A-layout (matches B): lane half 0 holds K 0..15,
    // half 1 holds K 16..31 for row m = lane&15.
    const int koffA = lhalf * 16;
    // B-layout used here (16-bit 32x16, matches A): lane half 0 holds K 0..15, half 1 K 16..31
    // for column n = lane&15 -> 16 contiguous halves from Bt[n][k+koffB].
    const int koffB = lhalf * 16;

    const _Float16* arow0 = A + (size_t)(m0 + lmod) * K;       // rows m0..m0+15
    const _Float16* arow1 = A + (size_t)(m0 + 16 + lmod) * K;  // rows m0+16..m0+31
    const _Float16* brow[4];
#pragma unroll
    for (int t = 0; t < 4; ++t)
        brow[t] = Bt + (size_t)(n0 + t * 16 + lmod) * K + koffB;

    v8f acc0_0 = {}, acc0_1 = {}, acc0_2 = {}, acc0_3 = {};
    v8f acc1_0 = {}, acc1_1 = {}, acc1_2 = {}, acc1_3 = {};

    for (int k = 0; k < K; k += 32) {
        v8h a0lo = *(const v8h*)(arow0 + k + koffA);
        v8h a0hi = *(const v8h*)(arow0 + k + 8 + koffA);
        v8h a1lo = *(const v8h*)(arow1 + k + koffA);
        v8h a1hi = *(const v8h*)(arow1 + k + 8 + koffA);
        v16h af0, af1;
#pragma unroll
        for (int i = 0; i < 8; ++i) {
            af0[i] = a0lo[i]; af0[i + 8] = a0hi[i];
            af1[i] = a1lo[i]; af1[i + 8] = a1hi[i];
        }
        v16h b0 = *(const v16h*)(brow[0] + k);
        v16h b1 = *(const v16h*)(brow[1] + k);
        v16h b2 = *(const v16h*)(brow[2] + k);
        v16h b3 = *(const v16h*)(brow[3] + k);
        acc0_0 = __builtin_amdgcn_wmma_f32_16x16x32_f16(false, af0, false, b0, (short)0, acc0_0, false, false);
        acc0_1 = __builtin_amdgcn_wmma_f32_16x16x32_f16(false, af0, false, b1, (short)0, acc0_1, false, false);
        acc0_2 = __builtin_amdgcn_wmma_f32_16x16x32_f16(false, af0, false, b2, (short)0, acc0_2, false, false);
        acc0_3 = __builtin_amdgcn_wmma_f32_16x16x32_f16(false, af0, false, b3, (short)0, acc0_3, false, false);
        acc1_0 = __builtin_amdgcn_wmma_f32_16x16x32_f16(false, af1, false, b0, (short)0, acc1_0, false, false);
        acc1_1 = __builtin_amdgcn_wmma_f32_16x16x32_f16(false, af1, false, b1, (short)0, acc1_1, false, false);
        acc1_2 = __builtin_amdgcn_wmma_f32_16x16x32_f16(false, af1, false, b2, (short)0, acc1_2, false, false);
        acc1_3 = __builtin_amdgcn_wmma_f32_16x16x32_f16(false, af1, false, b3, (short)0, acc1_3, false, false);
    }

    // C/D layout: VGPR r, lanes 0-15 -> (mtile+r, n0+lane); lanes 16-31 -> (mtile+8+r, ...)
#pragma unroll
    for (int a = 0; a < 2; ++a) {
        const int mbase = m0 + a * 16 + lhalf * 8;
#pragma unroll
        for (int r = 0; r < 8; ++r) {
            const int mm = mbase + r;
            float vr[4];
            if (a == 0) { vr[0] = acc0_0[r]; vr[1] = acc0_1[r]; vr[2] = acc0_2[r]; vr[3] = acc0_3[r]; }
            else        { vr[0] = acc1_0[r]; vr[1] = acc1_1[r]; vr[2] = acc1_2[r]; vr[3] = acc1_3[r]; }
#pragma unroll
            for (int t = 0; t < 4; ++t) {
                const int nn = n0 + t * 16 + lmod;
                float v = vr[t];
                if (bias) v += bias[nn];
                const size_t idx = (size_t)mm * N + nn;
                if (mode == 0) {
                    C[idx] = v;
                } else if (mode == 1) {
                    // exact GELU: 0.5*x*(1+erf(x/sqrt(2)))
                    C16[idx] = (_Float16)(0.5f * v * (1.0f + erff(v * 0.70710678118f)));
                } else {
                    C[idx] += v;
                }
            }
        }
    }
}

// --------------------------- elementwise helpers ---------------------------

__global__ void patchify_kernel(const float* __restrict__ X, _Float16* __restrict__ out)
{
    size_t t = (size_t)blockIdx.x * 256 + threadIdx.x;
    if (t >= (size_t)TOK_ * PD_) return;
    int pd = (int)(t % PD_);
    size_t tok = t / PD_;
    int n = (int)(tok % NTOK);
    int f = (int)((tok / NTOK) % F_);
    int b = (int)(tok / (NTOK * F_));
    int c  = pd % C_;
    int pp = pd / C_;
    int p1 = pp / P_, p2 = pp % P_;
    int hp = n / WP_, wp = n % WP_;
    size_t src = (((size_t)(b * F_ + f) * C_ + c) * H_ + (hp * P_ + p1)) * W_ + (wp * P_ + p2);
    out[t] = (_Float16)X[src];
}

__global__ void unpatchify_kernel(const float* __restrict__ y, float* __restrict__ out)
{
    size_t t = (size_t)blockIdx.x * 256 + threadIdx.x;
    if (t >= (size_t)B_ * F_ * C_ * H_ * W_) return;
    int ww = (int)(t % W_);
    size_t r = t / W_;
    int hh = (int)(r % H_); r /= H_;
    int c  = (int)(r % C_); r /= C_;
    int f  = (int)(r % F_);
    int b  = (int)(r / F_);
    int hp = hh >> 4, p1 = hh & 15, wp = ww >> 4, p2 = ww & 15;
    size_t tok = (size_t)(b * F_ + f) * NTOK + hp * WP_ + wp;
    out[t] = y[tok * PD_ + (size_t)(p1 * P_ + p2) * C_ + c];
}

// transpose-convert weight W[K,N] f32 -> Wt[N,K] f16 (K-contiguous for WMMA B)
__global__ void wconv_kernel(const float* __restrict__ W, _Float16* __restrict__ Wt, int K, int N)
{
    size_t t = (size_t)blockIdx.x * 256 + threadIdx.x;
    if (t >= (size_t)K * N) return;
    int n = (int)(t % N);
    int k = (int)(t / N);
    Wt[(size_t)n * K + k] = (_Float16)W[t];
}

__global__ void f32to16_kernel(const float* __restrict__ a, _Float16* __restrict__ o, size_t n)
{
    size_t t = (size_t)blockIdx.x * 256 + threadIdx.x;
    if (t < n) o[t] = (_Float16)a[t];
}

// ------------------------------ layernorm ----------------------------------
__global__ __launch_bounds__(256) void layernorm_kernel(
    const float* __restrict__ x, const float* __restrict__ g,
    const float* __restrict__ b, _Float16* __restrict__ out)
{
    const int row = blockIdx.x;
    const float* xr = x + (size_t)row * DIM_;
    __shared__ float red[256];
    float s = 0.f;
    for (int i = threadIdx.x; i < DIM_; i += 256) s += xr[i];
    red[threadIdx.x] = s; __syncthreads();
    for (int off = 128; off > 0; off >>= 1) {
        if (threadIdx.x < off) red[threadIdx.x] += red[threadIdx.x + off];
        __syncthreads();
    }
    const float mean = red[0] * (1.0f / DIM_);
    __syncthreads();
    float v = 0.f;
    for (int i = threadIdx.x; i < DIM_; i += 256) { float d = xr[i] - mean; v += d * d; }
    red[threadIdx.x] = v; __syncthreads();
    for (int off = 128; off > 0; off >>= 1) {
        if (threadIdx.x < off) red[threadIdx.x] += red[threadIdx.x + off];
        __syncthreads();
    }
    const float rstd = rsqrtf(red[0] * (1.0f / DIM_) + 1e-5f);
    for (int i = threadIdx.x; i < DIM_; i += 256)
        out[(size_t)row * DIM_ + i] = (_Float16)((xr[i] - mean) * rstd * g[i] + b[i]);
}

// ------------------------------ masks / rotary -----------------------------
__global__ void mask_kernel(const float* __restrict__ obs, float* __restrict__ obm)
{
    int t = blockIdx.x * 64 + threadIdx.x;            // B*F*NTOK = 3136
    if (t >= B_ * F_ * NTOK) return;
    int n = t % NTOK;
    int f = (t / NTOK) % F_;
    int b = t / (NTOK * F_);
    int hp = n / WP_, wp = n % WP_;
    float sum = 0.f;
    for (int c = 0; c < C_; ++c)
        for (int p1 = 0; p1 < P_; ++p1)
            for (int p2 = 0; p2 < P_; ++p2)
                sum += obs[(((size_t)(b * F_ + f) * C_ + c) * H_ + hp * P_ + p1) * W_ + wp * P_ + p2];
    obm[t] = (sum * (1.0f / PD_) > 0.8f) ? 1.0f : 0.0f;
}

__global__ void rot_tables_kernel(float* fsin, float* fcos, float* isin, float* icos)
{
    int t = blockIdx.x * 256 + threadIdx.x;
    if (t < F_ * DHEAD_) {                            // frame rotary
        int f = t >> 6, d = t & 63;
        float inv = powf(10000.f, -(float)(d >> 1) / 32.f);
        float fr = (float)f * inv;
        fsin[t] = sinf(fr); fcos[t] = cosf(fr);
    }
    int u = t - F_ * DHEAD_;
    if (u >= 0 && u < NTOK * DHEAD_) {                // image rotary
        int n = u >> 6, d = u & 63;
        int hp = n / WP_, wp = n % WP_;
        int tt = d >> 1;                              // 0..31
        float val;
        if (tt < 16) {
            float scale = 1.f + 4.f * (float)tt / 15.f;          // linspace(1,5,16)
            float pos   = -1.f + 2.f * (float)hp / (float)(HP_ - 1);
            val = pos * scale * PI_;
        } else {
            float scale = 1.f + 4.f * (float)(tt - 16) / 15.f;
            float pos   = -1.f + 2.f * (float)wp / (float)(WP_ - 1);
            val = pos * scale * PI_;
        }
        isin[u] = sinf(val); icos[u] = cosf(val);
    }
}

// ------------------------------ time attention -----------------------------
// One 64-thread block per (b, head, n): 8x8 attention over frames, dim 64.
__global__ __launch_bounds__(64) void time_attn_kernel(
    const float* __restrict__ qkv, const float* __restrict__ obm,
    const float* __restrict__ fsin, const float* __restrict__ fcos,
    _Float16* __restrict__ o16)
{
    int g = blockIdx.x;                               // B*HEADS*NTOK
    int n = g % NTOK; int t2 = g / NTOK;
    int h = t2 % HEADS_; int b = t2 / HEADS_;
    __shared__ float qs[F_][DHEAD_], ks[F_][DHEAD_], vs[F_][DHEAD_], sim[F_][F_ + 1];
    const int d = threadIdx.x;                        // 0..63

    for (int f = 0; f < F_; ++f) {
        size_t tok = (size_t)b * (F_ * NTOK) + (size_t)f * NTOK + n;
        const float* base = qkv + tok * (3 * INNER_) + h * DHEAD_;
        float qv = base[d] * SCALE_,         qp = base[d ^ 1] * SCALE_;
        float kv = base[INNER_ + d],         kp = base[INNER_ + (d ^ 1)];
        float rq = (d & 1) ? qp : -qp;                // rotate_every_two
        float rk = (d & 1) ? kp : -kp;
        float c = fcos[f * DHEAD_ + d], s = fsin[f * DHEAD_ + d];
        qs[f][d] = qv * c + rq * s;
        ks[f][d] = kv * c + rk * s;
        vs[f][d] = base[2 * INNER_ + d];
    }
    __syncthreads();

    {
        int i = d >> 3, j = d & 7;
        float s = 0.f;
        for (int dd = 0; dd < DHEAD_; ++dd) s += qs[i][dd] * ks[j][dd];
        if (i == j || obm[(b * F_ + j) * NTOK + n] == 0.f) s = NEGMAX;
        sim[i][j] = s;
    }
    __syncthreads();
    if (d < F_) {
        float mx = NEGMAX;
        for (int j = 0; j < F_; ++j) mx = fmaxf(mx, sim[d][j]);
        float e[F_], sum = 0.f;
        for (int j = 0; j < F_; ++j) { e[j] = expf(sim[d][j] - mx); sum += e[j]; }
        float inv = 1.f / sum;
        for (int j = 0; j < F_; ++j) sim[d][j] = e[j] * inv;
    }
    __syncthreads();
    for (int f = 0; f < F_; ++f) {
        float acc = 0.f;
        for (int j = 0; j < F_; ++j) acc += sim[f][j] * vs[j][d];
        size_t tok = (size_t)b * (F_ * NTOK) + (size_t)f * NTOK + n;
        o16[tok * INNER_ + h * DHEAD_ + d] = (_Float16)acc;
    }
}

// ------------------------------ space attention ----------------------------
// One 256-thread block per (b, head, f): 196x196 attention, dim 64.
// K,V staged in LDS as f16 (~50 KB), Q row rotated per iteration.
__global__ __launch_bounds__(256) void space_attn_kernel(
    const float* __restrict__ qkv, const float* __restrict__ obm,
    const float* __restrict__ isin, const float* __restrict__ icos,
    _Float16* __restrict__ o16)
{
    int g = blockIdx.x;                               // B*HEADS*F
    int h = g % HEADS_; int t2 = g / HEADS_;
    int f = t2 % F_; int b = t2 / F_;
    __shared__ _Float16 ks[NTOK][DHEAD_];
    __shared__ _Float16 vs[NTOK][DHEAD_];
    __shared__ float qrow[DHEAD_];
    __shared__ float simrow[256];
    __shared__ float red[256];
    const int tid = threadIdx.x;

    for (int idx = tid; idx < NTOK * DHEAD_; idx += 256) {
        int n = idx >> 6, d = idx & 63;
        size_t tok = (size_t)b * (F_ * NTOK) + (size_t)f * NTOK + n;
        const float* base = qkv + tok * (3 * INNER_) + h * DHEAD_;
        float kv = base[INNER_ + d], kp = base[INNER_ + (d ^ 1)];
        float rk = (d & 1) ? kp : -kp;
        ks[n][d] = (_Float16)(kv * icos[n * DHEAD_ + d] + rk * isin[n * DHEAD_ + d]);
        vs[n][d] = (_Float16)base[2 * INNER_ + d];
    }
    __syncthreads();

    for (int i = 0; i < NTOK; ++i) {
        size_t toki = (size_t)b * (F_ * NTOK) + (size_t)f * NTOK + i;
        if (tid < DHEAD_) {
            const float* base = qkv + toki * (3 * INNER_) + h * DHEAD_;
            float qv = base[tid] * SCALE_, qp = base[tid ^ 1] * SCALE_;
            float rq = (tid & 1) ? qp : -qp;
            qrow[tid] = qv * icos[i * DHEAD_ + tid] + rq * isin[i * DHEAD_ + tid];
        }
        __syncthreads();
        float s = NEGMAX;
        if (tid < NTOK) {
            s = 0.f;
            for (int d = 0; d < DHEAD_; ++d) s += qrow[d] * (float)ks[tid][d];
            if (tid == i || obm[(b * F_ + f) * NTOK + tid] == 0.f) s = NEGMAX;
        }
        simrow[tid] = s; red[tid] = s; __syncthreads();
        for (int off = 128; off > 0; off >>= 1) {
            if (tid < off) red[tid] = fmaxf(red[tid], red[tid + off]);
            __syncthreads();
        }
        float mx = red[0]; __syncthreads();
        float e = (tid < NTOK) ? expf(simrow[tid] - mx) : 0.f;
        simrow[tid] = e; red[tid] = e; __syncthreads();
        for (int off = 128; off > 0; off >>= 1) {
            if (tid < off) red[tid] += red[tid + off];
            __syncthreads();
        }
        float inv = 1.f / red[0]; __syncthreads();
        if (tid < DHEAD_) {
            float acc = 0.f;
            for (int j = 0; j < NTOK; ++j) acc += simrow[j] * (float)vs[j][tid];
            o16[toki * INNER_ + h * DHEAD_ + tid] = (_Float16)(acc * inv);
        }
        __syncthreads();
    }
}

// ---------------------------------------------------------------------------
extern "C" void kernel_launch(void* const* d_in, const int* in_sizes, int n_in,
                              void* d_out, int out_size, void* d_ws, size_t ws_size,
                              hipStream_t stream)
{
    (void)in_sizes; (void)n_in; (void)out_size; (void)ws_size;
    // setup_inputs() dict insertion order, nested depth-first:
    const float* X_in    = (const float*)d_in[0];
    const float* obs     = (const float*)d_in[1];
    const float* patch_w = (const float*)d_in[2];
    const float* patch_b = (const float*)d_in[3];
    const float* out_w   = (const float*)d_in[4];
    const float* out_b   = (const float*)d_in[5];
    const float* tln_g   = (const float*)d_in[6];
    const float* tln_b   = (const float*)d_in[7];
    const float* t_qkv   = (const float*)d_in[8];
    const float* t_ow    = (const float*)d_in[9];
    const float* t_ob    = (const float*)d_in[10];
    const float* sln_g   = (const float*)d_in[11];
    const float* sln_b   = (const float*)d_in[12];
    const float* s_qkv   = (const float*)d_in[13];
    const float* s_ow    = (const float*)d_in[14];
    const float* s_ob    = (const float*)d_in[15];
    const float* fln_g   = (const float*)d_in[16];
    const float* fln_b   = (const float*)d_in[17];
    const float* ff_w1   = (const float*)d_in[18];
    const float* ff_b1   = (const float*)d_in[19];
    const float* ff_w2   = (const float*)d_in[20];
    const float* ff_b2   = (const float*)d_in[21];

    // ---- workspace carve-up (~165 MB total) ----
    char* p = (char*)d_ws;
    auto take = [&](size_t bytes) {
        char* r = p; p += (bytes + 255) & ~(size_t)255; return r;
    };
    _Float16* patches16 = (_Float16*)take((size_t)TOK_ * PD_ * 2);
    float*    x         = (float*)   take((size_t)TOK_ * DIM_ * 4);
    _Float16* h16       = (_Float16*)take((size_t)TOK_ * DIM_ * 2);
    float*    qkvb      = (float*)   take((size_t)TOK_ * 3 * INNER_ * 4);
    _Float16* o16       = (_Float16*)take((size_t)TOK_ * INNER_ * 2);
    _Float16* ffh16     = (_Float16*)take((size_t)TOK_ * 4 * DIM_ * 2);
    _Float16* x16       = (_Float16*)take((size_t)TOK_ * DIM_ * 2);
    float*    y         = (float*)   take((size_t)TOK_ * PD_ * 4);
    float*    obm       = (float*)   take((size_t)B_ * F_ * NTOK * 4);
    float*    fsin      = (float*)   take(F_ * DHEAD_ * 4);
    float*    fcos      = (float*)   take(F_ * DHEAD_ * 4);
    float*    isin      = (float*)   take((size_t)NTOK * DHEAD_ * 4);
    float*    icos      = (float*)   take((size_t)NTOK * DHEAD_ * 4);
    _Float16* pw_t      = (_Float16*)take((size_t)PD_ * DIM_ * 2);
    _Float16* ow_t      = (_Float16*)take((size_t)DIM_ * PD_ * 2);
    _Float16* tqkv_t    = (_Float16*)take((size_t)DEPTH_ * DIM_ * 3 * INNER_ * 2);
    _Float16* tow_t     = (_Float16*)take((size_t)DEPTH_ * INNER_ * DIM_ * 2);
    _Float16* sqkv_t    = (_Float16*)take((size_t)DEPTH_ * DIM_ * 3 * INNER_ * 2);
    _Float16* sow_t     = (_Float16*)take((size_t)DEPTH_ * INNER_ * DIM_ * 2);
    _Float16* w1_t      = (_Float16*)take((size_t)DEPTH_ * DIM_ * 4 * DIM_ * 2);
    _Float16* w2_t      = (_Float16*)take((size_t)DEPTH_ * 4 * DIM_ * DIM_ * 2);

    auto nblk = [](size_t n) { return (unsigned)((n + 255) / 256); };
    auto gemm = [&](const _Float16* A, const _Float16* Bt, const float* bias,
                    float* C, _Float16* C16o, int M, int Nn, int K, int mode) {
        dim3 grid(M / 64, Nn / 128);
        gemm_wmma_kernel<<<grid, 128, 0, stream>>>(A, Bt, bias, C, C16o, M, Nn, K, mode);
    };
    auto wconv = [&](const float* W, _Float16* Wt, int K, int N) {
        wconv_kernel<<<nblk((size_t)K * N), 256, 0, stream>>>(W, Wt, K, N);
    };

    // ---- one-time prep: patches, masks, rotary tables, f16 weights ----
    patchify_kernel<<<nblk((size_t)TOK_ * PD_), 256, 0, stream>>>(X_in, patches16);
    mask_kernel<<<(B_ * F_ * NTOK) / 64, 64, 0, stream>>>(obs, obm);
    rot_tables_kernel<<<nblk(F_ * DHEAD_ + (size_t)NTOK * DHEAD_), 256, 0, stream>>>(fsin, fcos, isin, icos);

    wconv(patch_w, pw_t, PD_, DIM_);
    wconv(out_w,   ow_t, DIM_, PD_);
    for (int l = 0; l < DEPTH_; ++l) {
        wconv(t_qkv + (size_t)l * DIM_ * 3 * INNER_, tqkv_t + (size_t)l * DIM_ * 3 * INNER_, DIM_, 3 * INNER_);
        wconv(t_ow  + (size_t)l * INNER_ * DIM_,     tow_t  + (size_t)l * INNER_ * DIM_,     INNER_, DIM_);
        wconv(s_qkv + (size_t)l * DIM_ * 3 * INNER_, sqkv_t + (size_t)l * DIM_ * 3 * INNER_, DIM_, 3 * INNER_);
        wconv(s_ow  + (size_t)l * INNER_ * DIM_,     sow_t  + (size_t)l * INNER_ * DIM_,     INNER_, DIM_);
        wconv(ff_w1 + (size_t)l * DIM_ * 4 * DIM_,   w1_t   + (size_t)l * DIM_ * 4 * DIM_,   DIM_, 4 * DIM_);
        wconv(ff_w2 + (size_t)l * 4 * DIM_ * DIM_,   w2_t   + (size_t)l * 4 * DIM_ * DIM_,   4 * DIM_, DIM_);
    }

    // ---- patch embedding: x = patches @ patch_w + patch_b ----
    gemm(patches16, pw_t, patch_b, x, nullptr, TOK_, DIM_, PD_, 0);

    // ---- transformer layers ----
    for (int l = 0; l < DEPTH_; ++l) {
        // time attention
        layernorm_kernel<<<TOK_, 256, 0, stream>>>(x, tln_g + l * DIM_, tln_b + l * DIM_, h16);
        gemm(h16, tqkv_t + (size_t)l * DIM_ * 3 * INNER_, nullptr, qkvb, nullptr, TOK_, 3 * INNER_, DIM_, 0);
        time_attn_kernel<<<B_ * HEADS_ * NTOK, 64, 0, stream>>>(qkvb, obm, fsin, fcos, o16);
        gemm(o16, tow_t + (size_t)l * INNER_ * DIM_, t_ob + l * DIM_, x, nullptr, TOK_, DIM_, INNER_, 2);
        // space attention
        layernorm_kernel<<<TOK_, 256, 0, stream>>>(x, sln_g + l * DIM_, sln_b + l * DIM_, h16);
        gemm(h16, sqkv_t + (size_t)l * DIM_ * 3 * INNER_, nullptr, qkvb, nullptr, TOK_, 3 * INNER_, DIM_, 0);
        space_attn_kernel<<<B_ * HEADS_ * F_, 256, 0, stream>>>(qkvb, obm, isin, icos, o16);
        gemm(o16, sow_t + (size_t)l * INNER_ * DIM_, s_ob + l * DIM_, x, nullptr, TOK_, DIM_, INNER_, 2);
        // FFN
        layernorm_kernel<<<TOK_, 256, 0, stream>>>(x, fln_g + l * DIM_, fln_b + l * DIM_, h16);
        gemm(h16, w1_t + (size_t)l * DIM_ * 4 * DIM_, ff_b1 + (size_t)l * 4 * DIM_, nullptr, ffh16, TOK_, 4 * DIM_, DIM_, 1);
        gemm(ffh16, w2_t + (size_t)l * 4 * DIM_ * DIM_, ff_b2 + (size_t)l * DIM_, x, nullptr, TOK_, DIM_, 4 * DIM_, 2);
    }

    // ---- output projection + un-patchify ----
    f32to16_kernel<<<nblk((size_t)TOK_ * DIM_), 256, 0, stream>>>(x, x16, (size_t)TOK_ * DIM_);
    gemm(x16, ow_t, out_b, y, nullptr, TOK_, PD_, DIM_, 0);
    unpatchify_kernel<<<nblk((size_t)B_ * F_ * C_ * H_ * W_), 256, 0, stream>>>(y, (float*)d_out);
}
